// TimeDistanceViT_7035156431001
// MI455X (gfx1250) — matrix operand order repaired and checked by reading, not compiled, not run on hardware
//
#include <hip/hip_runtime.h>
#include <hip/hip_bf16.h>
#include <math.h>

// ---------------------------------------------------------------------------
// TimeDistanceViT forward on gfx1250 (MI455X).  wave32, WMMA f16->f32.
// ---------------------------------------------------------------------------

typedef _Float16 v16h __attribute__((ext_vector_type(16)));
typedef _Float16 v8h  __attribute__((ext_vector_type(8)));
typedef _Float16 v4h  __attribute__((ext_vector_type(4)));
typedef float    v8f  __attribute__((ext_vector_type(8)));

#define DEPTH    6
#define DIM      768
#define HEADS    12
#define DH       64
#define MLP_DIM  3072
#define NCLS     1000
#define NPATCH   196
#define NTOK     784            // T * NPATCH
#define BATCH    2
#define ROWS     (BATCH*NTOK)   // 1568
#define QKV3     (3*DIM)        // 2304

#define EPI_BIAS 1
#define EPI_RES  2
#define EPI_GELU 4
#define EPI_POS  8

union V16H { v16h v; _Float16 e[16]; };
union V8F  { v8f  v; float    e[8];  };

static __device__ __forceinline__ v8f wmma_f16(v16h a, v16h b, v8f c) {
  // v_wmma_f32_16x16x32_f16  (8-arg form, probe-confirmed)
  return __builtin_amdgcn_wmma_f32_16x16x32_f16(false, a, false, b, (short)0, c, false, false);
}

// A fragment 16x32 f16 from a row-major tile (stride in halves).
// element (m,k): lane = (m&15) + 16*((k>>3)&1); half h: k = (h&7) + 8*((lane>>4)&1) + 16*(h>>3)
static __device__ __forceinline__ v16h load_a_frag(const _Float16* base, int stride, int lane) {
  int m    = lane & 15;
  int ksel = (lane >> 4) & 1;
  const _Float16* row = base + m * stride + 8 * ksel;
  v8h lo = *(const v8h*)(row);        // k = 8*ksel + 0..7
  v8h hi = *(const v8h*)(row + 16);   // k = 16 + 8*ksel + 0..7
  return __builtin_shufflevector(lo, hi, 0,1,2,3,4,5,6,7,8,9,10,11,12,13,14,15);
}

// B fragment 32x16 f16 from a row-major (k-major) tile: n = lane&15; half h: k = (h&7) + 8*((lane>>4)&1) + 16*(h>>3).
static __device__ __forceinline__ v16h load_b_frag(const _Float16* base, int stride, int lane) {
  int n    = lane & 15;
  int ksel = (lane >> 4) & 1;
  V16H r;
#pragma unroll
  for (int i = 0; i < 8; i++) {
    r.e[i]     = base[(8 * ksel + i) * stride + n];
    r.e[8 + i] = base[(16 + 8 * ksel + i) * stride + n];
  }
  return r.v;
}

static __device__ __forceinline__ v4h cvt4(float4 f) {
  v4h h;
  h[0] = (_Float16)f.x; h[1] = (_Float16)f.y;
  h[2] = (_Float16)f.z; h[3] = (_Float16)f.w;
  return h;
}

// ---------------------------------------------------------------------------
// im2col:  img[b,t,c,224,224] -> patch[b*784+n][768]   (p1 p2 c ordering)
// ---------------------------------------------------------------------------
__global__ void im2col_kernel(const float* __restrict__ img, float* __restrict__ patch) {
  int e = blockIdx.x * 256 + threadIdx.x;
  if (e >= ROWS * DIM) return;
  int row = e / DIM, k = e % DIM;
  int b = row / NTOK, n = row % NTOK;
  int t = n / NPATCH, p = n % NPATCH;
  int gh = p / 14, gw = p % 14;
  int c = k % 3, q = k / 3;
  int p1 = q / 16, p2 = q % 16;
  size_t src = ((((size_t)(b * 4 + t) * 3 + c) * 224) + gh * 16 + p1) * 224 + gw * 16 + p2;
  patch[e] = img[src];
}

// ---------------------------------------------------------------------------
// Generic GEMM:  out[M,N] = A[M,K](f32) @ W[K,N](f32)  with f16 WMMA tiles.
// 256 threads = 8 waves; block tile 128x128; wave tile 32x64; K-step 32.
// Register-buffered pipeline: tile k+32 is loaded to VGPRs while WMMAs for
// tile k run out of LDS.  All staging is branchless (index clamping: OOB
// rows/cols only affect OOB outputs, which are never stored).
// ---------------------------------------------------------------------------
__global__ __launch_bounds__(256) void gemm_kernel(
    const float* __restrict__ A, const float* __restrict__ W,
    const float* __restrict__ bias, const float* __restrict__ res,
    const float* __restrict__ pos, float* __restrict__ out,
    int M, int N, int K, int epi) {
  __shared__ __align__(16) _Float16 As[128][40];   // 32 k + pad (16B-aligned rows)
  __shared__ __align__(16) _Float16 Bs[32][136];
  int tid = threadIdx.x, lane = tid & 31, wid = tid >> 5;
  int wm = wid & 3, wn = wid >> 2;                 // wave grid 4(M) x 2(N)
  int bm = blockIdx.y * 128, bn = blockIdx.x * 128;
  bool fullM = (bm + 128 <= M);
  bool fullN = (bn + 128 <= N);

  v8f acc[2][4];
#pragma unroll
  for (int i = 0; i < 2; i++)
#pragma unroll
    for (int j = 0; j < 4; j++) acc[i][j] = (v8f)0.0f;

  float4 ar[4], br[4];

  auto load_tiles = [&](int k0) {
    if (fullM) {
#pragma unroll
      for (int i = 0; i < 4; i++) {                // A tile 128x32, float4
        int idx = tid + i * 256;
        int r = idx >> 3, c4 = (idx & 7) * 4;
        ar[i] = *(const float4*)(A + (size_t)(bm + r) * K + k0 + c4);
      }
    } else {
#pragma unroll
      for (int i = 0; i < 4; i++) {
        int idx = tid + i * 256;
        int r = min(bm + (idx >> 3), M - 1);       // clamp: branchless
        int c4 = (idx & 7) * 4;
        const float* p = A + (size_t)r * K + k0 + c4;
        ar[i] = make_float4(p[0], p[1], p[2], p[3]);
      }
    }
    if (fullN) {
#pragma unroll
      for (int i = 0; i < 4; i++) {                // B tile 32x128, float4
        int idx = tid + i * 256;
        int r = idx >> 5, c4 = (idx & 31) * 4;
        br[i] = *(const float4*)(W + (size_t)(k0 + r) * N + bn + c4);
      }
    } else {
#pragma unroll
      for (int i = 0; i < 4; i++) {
        int idx = tid + i * 256;
        int r = idx >> 5, c4 = (idx & 31) * 4;
        const float* rowp = W + (size_t)(k0 + r) * N;
        float4 f;
        f.x = rowp[min(bn + c4 + 0, N - 1)];
        f.y = rowp[min(bn + c4 + 1, N - 1)];
        f.z = rowp[min(bn + c4 + 2, N - 1)];
        f.w = rowp[min(bn + c4 + 3, N - 1)];
        br[i] = f;
      }
    }
  };

  load_tiles(0);
  for (int k0 = 0; k0 < K; k0 += 32) {
#pragma unroll
    for (int i = 0; i < 4; i++) {                  // convert + store to LDS
      int idx = tid + i * 256;
      *(v4h*)(&As[idx >> 3][(idx & 7) * 4]) = cvt4(ar[i]);
    }
#pragma unroll
    for (int i = 0; i < 4; i++) {
      int idx = tid + i * 256;
      *(v4h*)(&Bs[idx >> 5][(idx & 31) * 4]) = cvt4(br[i]);
    }
    __syncthreads();
    if (k0 + 32 < K) load_tiles(k0 + 32);          // overlap with WMMAs below
    if (k0 + 64 < K && tid < 64)                   // gfx1250 global_prefetch_b8
      __builtin_prefetch(W + (size_t)(k0 + 64 + (tid >> 1)) * N + bn + (tid & 1) * 64, 0, 0);
    v16h af[2], bf[4];
#pragma unroll
    for (int mi = 0; mi < 2; mi++) af[mi] = load_a_frag(&As[wm * 32 + mi * 16][0], 40, lane);
#pragma unroll
    for (int ni = 0; ni < 4; ni++) bf[ni] = load_b_frag(&Bs[0][wn * 64 + ni * 16], 136, lane);
#pragma unroll
    for (int mi = 0; mi < 2; mi++)
#pragma unroll
      for (int ni = 0; ni < 4; ni++) acc[mi][ni] = wmma_f16(af[mi], bf[ni], acc[mi][ni]);
    __syncthreads();
  }

  int hi = lane >> 4, nn = lane & 15;              // C/D: lane = n + 16*(m>>3), vgpr = m&7
#pragma unroll
  for (int mi = 0; mi < 2; mi++)
#pragma unroll
    for (int ni = 0; ni < 4; ni++) {
      V8F a; a.v = acc[mi][ni];
#pragma unroll
      for (int r = 0; r < 8; r++) {
        int row = bm + wm * 32 + mi * 16 + hi * 8 + r;
        int col = bn + wn * 64 + ni * 16 + nn;
        if (row < M && col < N) {
          float v = a.e[r];
          if (epi & EPI_BIAS) v += bias[col];
          if (epi & EPI_POS)  v += pos[(size_t)(row % NPATCH) * N + col];
          if (epi & EPI_GELU) v = 0.5f * v * (1.0f + erff(v * 0.70710678118f));
          if (epi & EPI_RES)  v += res[(size_t)row * N + col];
          out[(size_t)row * N + col] = v;
        }
      }
    }
}

// ---------------------------------------------------------------------------
// LayerNorm: one wave per row of 768; output f32.
// ---------------------------------------------------------------------------
__global__ __launch_bounds__(256) void ln_kernel(
    const float* __restrict__ x, const float* __restrict__ w,
    const float* __restrict__ b, float* __restrict__ out, int rows) {
  int lane = threadIdx.x & 31, wid = threadIdx.x >> 5;
  int row = blockIdx.x * 8 + wid;
  if (row >= rows) return;
  const float* xr = x + (size_t)row * DIM;
  float vals[24], s = 0.0f;
#pragma unroll
  for (int i = 0; i < 24; i++) { vals[i] = xr[lane + 32 * i]; s += vals[i]; }
#pragma unroll
  for (int off = 16; off; off >>= 1) s += __shfl_xor(s, off, 32);
  float mean = s * (1.0f / DIM), v = 0.0f;
#pragma unroll
  for (int i = 0; i < 24; i++) { float d = vals[i] - mean; v += d * d; }
#pragma unroll
  for (int off = 16; off; off >>= 1) v += __shfl_xor(v, off, 32);
  float inv = rsqrtf(v * (1.0f / DIM) + 1e-5f);
  float* o = out + (size_t)row * DIM;
#pragma unroll
  for (int i = 0; i < 24; i++) {
    int c = lane + 32 * i;
    o[c] = (vals[i] - mean) * inv * w[c] + b[c];
  }
}

// ---------------------------------------------------------------------------
// Attention: grid = B*H*7 blocks of 7 waves (224 thr); each wave owns exactly
// one 16-row tile (7*7 = 49 tiles per (b,h)).  K^T staged per block in LDS
// (re-staged per group: ~33MB extra HBM traffic total, ~1.5us @ 23.3TB/s,
// bought 7x more concurrency).  Per-row scalar decay; online (flash) softmax;
// P accum->A-frag re-layout via per-wave LDS bounce.  All global access is
// branchless (clamped); masking happens in the softmax (P==0 for j>=784).
// ---------------------------------------------------------------------------
__global__ __launch_bounds__(224) void attn_kernel(
    const float* __restrict__ qkv, const float* __restrict__ times,
    const float* __restrict__ ta_a, const float* __restrict__ ta_c,
    float* __restrict__ obuf) {
  __shared__ __align__(16) _Float16 Kt[64][800];    // K^T (cols j>=784 clamped)
  __shared__ __align__(16) _Float16 Pbuf[7][16 * 32];
  int g  = blockIdx.x % 7;                          // row-tile group
  int bh = blockIdx.x / 7;
  int b = bh / HEADS, h = bh % HEADS;
  int tid = threadIdx.x, lane = tid & 31, wid = tid >> 5;   // wid in 0..6
  const float* base = qkv + (size_t)b * NTOK * QKV3;

  for (int idx = tid; idx < 800 * 64; idx += 224) { // d fast: coalesced global
    int j = idx >> 6, d = idx & 63;
    int jc = min(j, NTOK - 1);
    Kt[d][j] = (_Float16)base[(size_t)jc * QKV3 + DIM + h * DH + d];
  }
  __syncthreads();

  float aa = fabsf(ta_a[h]), cc = fabsf(ta_c[h]);
  float t0 = times[b * 4];
  const float scale = 0.125f;                       // DH^-0.5
  int hi = lane >> 4, nn = lane & 15;

  int rt = g * 7 + wid;                             // 0..48, one tile per wave
  int i0 = rt * 16;
  v16h qa[2];
  {
    const float* qp = base + (size_t)(i0 + nn) * QKV3 + h * DH;
#pragma unroll
    for (int c = 0; c < 2; c++) {
      const float* p = qp + c * 32 + 8 * hi;
      V16H a;
#pragma unroll
      for (int i = 0; i < 8; i++) {
        a.e[i]     = (_Float16)(p[i] * scale);
        a.e[8 + i] = (_Float16)(p[16 + i] * scale);
      }
      qa[c] = a.v;
    }
  }
  float dec[8];
#pragma unroll
  for (int r = 0; r < 8; r++) {                     // decay is per-row only
    int row = i0 + hi * 8 + r;
    float R = fabsf(t0 - times[b * 4 + row / NPATCH]);
    dec[r] = 1.0f / (1.0f + __expf(aa * R - cc));
  }
  v8f o[4];
#pragma unroll
  for (int nt = 0; nt < 4; nt++) o[nt] = (v8f)0.0f;
  float rmax[8], rsum[8];
#pragma unroll
  for (int r = 0; r < 8; r++) { rmax[r] = 0.0f; rsum[r] = 0.0f; } // scores >= 0

  for (int ch = 0; ch < 25; ch++) {                 // 25 chunks of 32 cols
    int j0 = ch * 32;
    v8f s0 = (v8f)0.0f, s1 = (v8f)0.0f;
    s0 = wmma_f16(qa[0], load_b_frag(&Kt[0][j0], 800, lane), s0);
    s0 = wmma_f16(qa[1], load_b_frag(&Kt[32][j0], 800, lane), s0);
    s1 = wmma_f16(qa[0], load_b_frag(&Kt[0][j0 + 16], 800, lane), s1);
    s1 = wmma_f16(qa[1], load_b_frag(&Kt[32][j0 + 16], 800, lane), s1);
    // issue V loads for this chunk early; they drain while softmax runs
    float4 vf[16];
    {
      const float* vp = base + 2 * DIM + h * DH + nn;
#pragma unroll
      for (int gg = 0; gg < 16; gg++) {
        int nt = gg >> 2, s = (gg & 3) * 4;
        int k = (s & 7) + 8 * hi + 16 * (s >> 3);
        const float* q = vp + nt * 16;
        vf[gg].x = q[(size_t)min(j0 + k + 0, NTOK - 1) * QKV3];   // clamp: branchless
        vf[gg].y = q[(size_t)min(j0 + k + 1, NTOK - 1) * QKV3];
        vf[gg].z = q[(size_t)min(j0 + k + 2, NTOK - 1) * QKV3];
        vf[gg].w = q[(size_t)min(j0 + k + 3, NTOK - 1) * QKV3];
      }
    }
    V8F u0, u1; u0.v = s0; u1.v = s1;
    bool vl0 = (j0 + nn) < NTOK, vl1 = (j0 + 16 + nn) < NTOK;
    float p0a[8], p1a[8], corr[8];
#pragma unroll
    for (int r = 0; r < 8; r++) {
      float e0 = fmaxf(u0.e[r], 0.0f) * dec[r];
      float e1 = fmaxf(u1.e[r], 0.0f) * dec[r];
      float tmax = fmaxf(vl0 ? e0 : 0.0f, vl1 ? e1 : 0.0f);
#pragma unroll
      for (int m = 1; m < 16; m <<= 1) tmax = fmaxf(tmax, __shfl_xor(tmax, m, 32));
      float nm = fmaxf(rmax[r], tmax);
      float c  = __expf(rmax[r] - nm);
      float p0 = vl0 ? __expf(e0 - nm) : 0.0f;
      float p1 = vl1 ? __expf(e1 - nm) : 0.0f;
      float ts = p0 + p1;
#pragma unroll
      for (int m = 1; m < 16; m <<= 1) ts += __shfl_xor(ts, m, 32);
      rsum[r] = rsum[r] * c + ts;
      rmax[r] = nm;
      corr[r] = c; p0a[r] = p0; p1a[r] = p1;
    }
#pragma unroll
    for (int nt = 0; nt < 4; nt++) {
      V8F w; w.v = o[nt];
#pragma unroll
      for (int r = 0; r < 8; r++) w.e[r] *= corr[r];
      o[nt] = w.v;
    }
    _Float16* pb = Pbuf[wid];                       // accum-layout -> A-frag via LDS
#pragma unroll
    for (int r = 0; r < 8; r++) {
      pb[(hi * 8 + r) * 32 + nn]      = (_Float16)p0a[r];
      pb[(hi * 8 + r) * 32 + 16 + nn] = (_Float16)p1a[r];
    }
    asm volatile("s_wait_dscnt 0" ::: "memory");
    v16h pa = load_a_frag(pb, 32, lane);
#pragma unroll
    for (int nt = 0; nt < 4; nt++) {
      V16H vb;
#pragma unroll
      for (int gg = 0; gg < 4; gg++) {
        float4 f = vf[nt * 4 + gg];
        vb.e[gg*4+0] = (_Float16)f.x; vb.e[gg*4+1] = (_Float16)f.y;
        vb.e[gg*4+2] = (_Float16)f.z; vb.e[gg*4+3] = (_Float16)f.w;
      }
      o[nt] = wmma_f16(pa, vb.v, o[nt]);
    }
  }
  float* ob = obuf + (size_t)b * NTOK * DIM;
#pragma unroll
  for (int nt = 0; nt < 4; nt++) {
    V8F w; w.v = o[nt];
#pragma unroll
    for (int r = 0; r < 8; r++) {
      int row = i0 + hi * 8 + r;
      ob[(size_t)row * DIM + h * DH + nt * 16 + nn] = w.e[r] / rsum[r];
    }
  }
}

// ---------------------------------------------------------------------------
// Mean pool over tokens.
// ---------------------------------------------------------------------------
__global__ void pool_kernel(const float* __restrict__ X, float* __restrict__ out) {
  int e = blockIdx.x * 256 + threadIdx.x;
  if (e >= BATCH * DIM) return;
  int b = e / DIM, d = e % DIM;
  float s = 0.0f;
  for (int n = 0; n < NTOK; n++) s += X[((size_t)b * NTOK + n) * DIM + d];
  out[e] = s * (1.0f / NTOK);
}

// ---------------------------------------------------------------------------
static void launch_gemm(hipStream_t st, const float* A, const float* W,
                        const float* bias, const float* res, const float* pos,
                        float* out, int M, int N, int K, int epi) {
  dim3 g((N + 127) / 128, (M + 127) / 128);
  gemm_kernel<<<g, 256, 0, st>>>(A, W, bias, res, pos, out, M, N, K, epi);
}

extern "C" void kernel_launch(void* const* d_in, const int* in_sizes, int n_in,
                              void* d_out, int out_size, void* d_ws, size_t ws_size,
                              hipStream_t stream) {
  const float* img     = (const float*)d_in[0];
  const float* times   = (const float*)d_in[1];
  const float* patch_w = (const float*)d_in[2];
  const float* patch_b = (const float*)d_in[3];
  const float* pos_emb = (const float*)d_in[4];
  const float* ln1_w   = (const float*)d_in[5];
  const float* ln1_b   = (const float*)d_in[6];
  const float* wqkv    = (const float*)d_in[7];
  const float* wout    = (const float*)d_in[8];
  const float* ta_a    = (const float*)d_in[9];
  const float* ta_c    = (const float*)d_in[10];
  const float* ln2_w   = (const float*)d_in[11];
  const float* ln2_b   = (const float*)d_in[12];
  const float* w1      = (const float*)d_in[13];
  const float* b1      = (const float*)d_in[14];
  const float* w2      = (const float*)d_in[15];
  const float* b2      = (const float*)d_in[16];
  const float* hln_w   = (const float*)d_in[17];
  const float* hln_b   = (const float*)d_in[18];
  const float* hw1     = (const float*)d_in[19];
  const float* hb1     = (const float*)d_in[20];
  const float* hw2     = (const float*)d_in[21];
  const float* hb2     = (const float*)d_in[22];

  // workspace layout (floats); HID aliases QKV+OBUF (3072 = 2304 + 768)
  float* X     = (float*)d_ws;
  float* XN    = X   + (size_t)ROWS * DIM;
  float* QKV   = XN  + (size_t)ROWS * DIM;
  float* OBUF  = QKV + (size_t)ROWS * QKV3;
  float* HID   = QKV;
  float* PATCH = QKV;
  float* POOL  = OBUF + (size_t)ROWS * DIM;
  float* POOLN = POOL + BATCH * DIM;
  float* H1    = POOLN + BATCH * DIM;

  im2col_kernel<<<(ROWS * DIM + 255) / 256, 256, 0, stream>>>(img, PATCH);
  launch_gemm(stream, PATCH, patch_w, patch_b, nullptr, pos_emb, X,
              ROWS, DIM, DIM, EPI_BIAS | EPI_POS);

  for (int l = 0; l < DEPTH; l++) {
    ln_kernel<<<(ROWS + 7) / 8, 256, 0, stream>>>(X, ln1_w + l * DIM, ln1_b + l * DIM, XN, ROWS);
    launch_gemm(stream, XN, wqkv + (size_t)l * DIM * QKV3, nullptr, nullptr, nullptr,
                QKV, ROWS, QKV3, DIM, 0);
    attn_kernel<<<BATCH * HEADS * 7, 224, 0, stream>>>(QKV, times, ta_a + l * HEADS,
                                                       ta_c + l * HEADS, OBUF);
    launch_gemm(stream, OBUF, wout + (size_t)l * DIM * DIM, nullptr, X, nullptr,
                X, ROWS, DIM, DIM, EPI_RES);
    ln_kernel<<<(ROWS + 7) / 8, 256, 0, stream>>>(X, ln2_w + l * DIM, ln2_b + l * DIM, XN, ROWS);
    launch_gemm(stream, XN, w1 + (size_t)l * DIM * MLP_DIM, b1 + l * MLP_DIM, nullptr, nullptr,
                HID, ROWS, MLP_DIM, DIM, EPI_BIAS | EPI_GELU);
    launch_gemm(stream, HID, w2 + (size_t)l * MLP_DIM * DIM, b2 + l * DIM, X, nullptr,
                X, ROWS, DIM, MLP_DIM, EPI_BIAS | EPI_RES);
  }

  pool_kernel<<<(BATCH * DIM + 255) / 256, 256, 0, stream>>>(X, POOL);
  ln_kernel<<<1, 256, 0, stream>>>(POOL, hln_w, hln_b, POOLN, BATCH);
  launch_gemm(stream, POOLN, hw1, hb1, nullptr, nullptr, H1, BATCH, DIM, DIM, EPI_BIAS);
  launch_gemm(stream, H1, hw2, hb2, nullptr, nullptr, (float*)d_out, BATCH, NCLS, DIM, EPI_BIAS);
}
